// _Transcriber_34565896798554
// MI455X (gfx1250) — matrix operand order repaired and checked by reading, not compiled, not run on hardware
//
#include <hip/hip_runtime.h>
#include <hip/hip_bf16.h>
#include <math.h>

// ---------------- model constants ----------------
#define NB    8      // batch
#define TIN   1040
#define INF_  80     // input feature
#define TRI   128
#define DM    512    // model dim
#define FFN_  2048
#define NL    8      // layers
#define OUTD  1024
#define NH    8      // heads
#define DH    64     // head dim
#define QLEN  336    // Rt + U + nseg = 64 + 256 + 16
#define KLEN  335    // (nseg-1) + Rt + U = 15 + 64 + 256
#define RT    64
#define UU    256
#define NSEG  16
#define TR    260    // U + RC after stride fold
#define BD    (NB*DM)   // 4096, row stride of (T, B, D) buffers

// ---------------- WMMA plumbing ----------------
typedef __attribute__((ext_vector_type(16))) __bf16 v16bf;
typedef __attribute__((ext_vector_type(8)))  float  v8f;
typedef __attribute__((ext_vector_type(4)))  unsigned int v4u;
typedef __attribute__((ext_vector_type(4)))  float v4f;

union FragB { v16bf v; unsigned short u[16]; v4u q[2]; };
union FragC { v8f v; float f[8]; };

__device__ __forceinline__ unsigned short f2bf(float x) {
  union { float f; unsigned u; } c; c.f = x;
  return (unsigned short)(c.u >> 16);
}

__device__ __forceinline__ v8f wmma_bf16(const FragB& a, const FragB& b, v8f c) {
  return __builtin_amdgcn_wmma_f32_16x16x32_bf16(false, a.v, false, b.v,
                                                 (short)0, c, false, false);
}

// Fragments staged in LDS in lane-major order: lane l's 16 bf16 operand
// values live at base + l*16, contiguous -> two ds_load_b128 per fragment.
__device__ __forceinline__ FragB load_frag(const unsigned short* base, int lane) {
  FragB f;
  const v4u* p = (const v4u*)(base + (lane << 4));
  f.q[0] = p[0];
  f.q[1] = p[1];
  return f;
}

// Pack 8 floats -> 8 bf16 -> one 16B LDS store.
__device__ __forceinline__ void store8(unsigned short* dst, const float* v) {
  v4u r;
  r[0] = (unsigned)f2bf(v[0]) | ((unsigned)f2bf(v[1]) << 16);
  r[1] = (unsigned)f2bf(v[2]) | ((unsigned)f2bf(v[3]) << 16);
  r[2] = (unsigned)f2bf(v[4]) | ((unsigned)f2bf(v[5]) << 16);
  r[3] = (unsigned)f2bf(v[6]) | ((unsigned)f2bf(v[7]) << 16);
  *(v4u*)dst = r;
}

// Fragment-layout index helpers (16-bit WMMA operands, wave32):
//  A 16x32 tile: lane = row + 16*((k>>4)&1), slot j = k & 15
//  B 32x16 tile: lane = col + 16*((k>>4)&1), slot j = k & 15

// ---------------- generic bf16-WMMA GEMM ----------------
// C[M,N] = act(A[M,K] @ W[K,N] + bias) (+ resid).  act: 0=none, 1=relu.
// Block: 256 threads = 8 waves (2 x 4), 64x128 tile; each wave owns 32x32.
// All fills are branch-free: clamped addresses + cndmask zeroing.
__global__ void gemm_wmma_kernel(const float* __restrict__ A,
                                 const float* __restrict__ W,
                                 const float* __restrict__ bias,
                                 const float* __restrict__ resid,
                                 float* __restrict__ C,
                                 int M, int N, int K, int act) {
  __shared__ __align__(16) unsigned short Asf[4 * 512];  // 4 A-frag chunks (16 rows)
  __shared__ __align__(16) unsigned short Wsf[8 * 512];  // 8 B-frag chunks (16 cols)

  const int tid  = threadIdx.x;
  const int lane = tid & 31;
  const int wave = tid >> 5;
  const int wm   = wave >> 2;   // 0..1  (row group of 32)
  const int wn   = wave & 3;    // 0..3  (col group of 32)
  const int m0   = blockIdx.x * 64;
  const int n0   = blockIdx.y * 128;

  // A fill geometry (per thread, loop-invariant)
  const int ar  = tid >> 2;
  const int ac0 = (tid & 3) * 8;
  const int agm = m0 + ar;
  const bool amok = agm < M;
  const size_t arow = (size_t)(amok ? agm : (M - 1)) * K;
  unsigned short* adst =
      &Asf[(ar >> 4) * 512 +
           ((ar & 15) + 16 * ((ac0 >> 4) & 1)) * 16 + (ac0 & 8)];

  FragC acc[2][2];
#pragma unroll
  for (int i = 0; i < 2; ++i)
#pragma unroll
    for (int jx = 0; jx < 2; ++jx)
#pragma unroll
      for (int r = 0; r < 8; ++r) acc[i][jx].f[r] = 0.f;

  for (int kk = 0; kk < K; kk += 32) {
    if (kk + 32 < K)
      __builtin_prefetch(&A[arow + kk + 32], 0, 1);

    // ---- A tile: 64x32, one 8-wide run per thread (K runs are 8-aligned,
    //      K % 8 == 0 for every GEMM in this model -> run all-in or all-out)
    {
      const bool kok = (kk + ac0 + 8 <= K);
      const bool ok  = amok && kok;
      const v4f* p = (const v4f*)(A + arow + (kok ? (kk + ac0) : 0));
      v4f x0 = p[0], x1 = p[1];
      float vals[8];
#pragma unroll
      for (int j = 0; j < 4; ++j) {
        vals[j]     = ok ? x0[j] : 0.f;
        vals[4 + j] = ok ? x1[j] : 0.f;
      }
      store8(adst, vals);
    }

    // ---- W tile: 32x128, gathered along K (lane-coalesced b32 loads,
    //      contiguous frag-order b128 stores), branch-free ----
#pragma unroll
    for (int t = 0; t < 2; ++t) {
      int task = tid + t * 256;        // 0..511
      int nn   = task & 127;
      int k0   = (task >> 7) * 8;      // 0,8,16,24
      int gn   = n0 + nn;
      const bool nok = gn < N;
      const bool kok = (kk + k0 + 8 <= K);
      const bool ok  = nok && kok;
      const int gnc  = nok ? gn : (N - 1);
      const int gk0  = kok ? (kk + k0) : 0;
      float vals[8];
#pragma unroll
      for (int j = 0; j < 8; ++j)
        vals[j] = W[(size_t)(gk0 + j) * N + gnc];
#pragma unroll
      for (int j = 0; j < 8; ++j)
        vals[j] = ok ? vals[j] : 0.f;
      int lane_d = (nn & 15) + 16 * ((k0 >> 4) & 1);
      store8(&Wsf[(nn >> 4) * 512 + lane_d * 16 + (k0 & 8)], vals);
    }
    __syncthreads();

    FragB a0 = load_frag(Asf + (wm * 2    ) * 512, lane);
    FragB a1 = load_frag(Asf + (wm * 2 + 1) * 512, lane);
    FragB b0 = load_frag(Wsf + (wn * 2    ) * 512, lane);
    FragB b1 = load_frag(Wsf + (wn * 2 + 1) * 512, lane);
    acc[0][0].v = wmma_bf16(a0, b0, acc[0][0].v);
    acc[0][1].v = wmma_bf16(a0, b1, acc[0][1].v);
    acc[1][0].v = wmma_bf16(a1, b0, acc[1][0].v);
    acc[1][1].v = wmma_bf16(a1, b1, acc[1][1].v);
    __syncthreads();
  }

  const int mrow0 = (lane < 16) ? 0 : 8;
#pragma unroll
  for (int i = 0; i < 2; ++i)
#pragma unroll
    for (int jx = 0; jx < 2; ++jx) {
      int n = n0 + wn * 32 + jx * 16 + (lane & 15);
      if (n >= N) continue;
      float bv = bias ? bias[n] : 0.f;
#pragma unroll
      for (int r = 0; r < 8; ++r) {
        int m = m0 + wm * 32 + i * 16 + mrow0 + r;
        if (m >= M) continue;
        float v = acc[i][jx].f[r] + bv;
        if (act == 1) v = fmaxf(v, 0.f);
        if (resid) v += resid[(size_t)m * N + n];
        C[(size_t)m * N + n] = v;
      }
    }
}

// ---------------- attention (one wave per block) ----------------
// grid = (NB*NH, QLEN/16), block = 32. q/k/v buffers are (T, B, D) with head h
// at columns [h*DH, h*DH+DH).
__global__ void attention_kernel(const float* __restrict__ qbuf,
                                 const float* __restrict__ kbuf,
                                 const float* __restrict__ vbuf,
                                 const unsigned char* __restrict__ mask,
                                 const int* __restrict__ lengths,
                                 float* __restrict__ obuf) {
  __shared__ __align__(16) unsigned short qsf[2 * 512];   // q A-frags (dh chunks)
  __shared__ __align__(16) unsigned short ksf[2 * 512];   // k^T B-frags (dh chunks)
  __shared__ __align__(16) unsigned short vsf[4 * 512];   // V B-frags (dh tiles)
  __shared__ __align__(16) unsigned short psf[11 * 512];  // probs A-frags (kt chunks)
  __shared__ float sc[16 * 336];
  __shared__ float rowm[16], rowinv[16];

  const int bh   = blockIdx.x;
  const int b    = bh / NH;
  const int h    = bh % NH;
  const int q0   = blockIdx.y * 16;
  const int lane = threadIdx.x;

  int maxlr = 0;
#pragma unroll
  for (int i = 0; i < NB; ++i) maxlr = max(maxlr, lengths[i] / 4);
  const int klen = lengths[b] / 4 + (NSEG - 1) + (QLEN - maxlr - NSEG);

  const size_t headoff = (size_t)b * DM + h * DH;

  // q tile (16x64, scaled by DH^-0.5 = 0.125), stored as A-fragments
  for (int task = lane; task < 128; task += 32) {
    int r = task >> 3, c0 = (task & 7) * 8;
    const v4f* p = (const v4f*)(qbuf + (size_t)(q0 + r) * BD + headoff + c0);
    v4f x0 = p[0], x1 = p[1];
    float vals[8];
#pragma unroll
    for (int j = 0; j < 4; ++j) {
      vals[j]     = x0[j] * 0.125f;
      vals[4 + j] = x1[j] * 0.125f;
    }
    int chunk = c0 >> 5, cw = c0 & 31;
    int lane_d = r + 16 * ((cw >> 4) & 1);
    int j0 = cw & 8;
    store8(&qsf[chunk * 512 + lane_d * 16 + j0], vals);
  }
  __syncthreads();
  FragB a0 = load_frag(qsf, lane);
  FragB a1 = load_frag(qsf + 512, lane);

  // pass 1: scores = q @ k^T, masked, into sc
  for (int kt = 0; kt < KLEN; kt += 16) {
    for (int task = lane; task < 128; task += 32) {
      int r = task >> 3, c0 = (task & 7) * 8;
      int kr = kt + r;
      const bool ok = kr < KLEN;
      const int krc = ok ? kr : (KLEN - 1);
      const v4f* p = (const v4f*)(kbuf + (size_t)krc * BD + headoff + c0);
      v4f x0 = p[0], x1 = p[1];
      float vals[8];
#pragma unroll
      for (int j = 0; j < 4; ++j) {
        vals[j]     = ok ? x0[j] : 0.f;
        vals[4 + j] = ok ? x1[j] : 0.f;
      }
      // B^T fragment: lane = k-row, slot = dh & 15
      int chunk = c0 >> 5, cw = c0 & 31;
      int lane_d = r + 16 * ((cw >> 4) & 1);
      store8(&ksf[chunk * 512 + lane_d * 16 + (cw & 8)], vals);
    }
    __syncthreads();
    FragB b0 = load_frag(ksf, lane);
    FragB b1 = load_frag(ksf + 512, lane);
    FragC c;
#pragma unroll
    for (int i = 0; i < 8; ++i) c.f[i] = 0.f;
    c.v = wmma_bf16(a0, b0, c.v);
    c.v = wmma_bf16(a1, b1, c.v);

    const int n = lane & 15;
    const int mrow0 = (lane < 16) ? 0 : 8;
#pragma unroll
    for (int r = 0; r < 8; ++r) {
      int qrow = q0 + mrow0 + r;
      int kcol = kt + n;
      float v = c.f[r];
      if (kcol >= KLEN || kcol >= klen || mask[(size_t)qrow * KLEN + kcol])
        v = -1.0e8f;
      sc[(mrow0 + r) * 336 + kcol] = v;
    }
    __syncthreads();
  }

  // softmax: lane pair (2*row, 2*row+1) reduces one row; broadcast m/inv
  {
    int row = lane >> 1, par = lane & 1;
    float m = -1.0e30f;
    for (int c = par; c < 336; c += 2) m = fmaxf(m, sc[row * 336 + c]);
    m = fmaxf(m, __shfl_xor(m, 1));
    float s = 0.f;
    for (int c = par; c < 336; c += 2) s += __expf(sc[row * 336 + c] - m);
    s += __shfl_xor(s, 1);
    if (par == 0) { rowm[row] = m; rowinv[row] = 1.f / s; }
  }
  __syncthreads();

  // probs -> psf as A-fragments (8-wide packed b128 stores)
  for (int task = lane; task < 16 * 44; task += 32) {
    int r = task / 44, g = task % 44;
    int c0 = g * 8;
    float m = rowm[r], inv = rowinv[r];
    float vals[8];
#pragma unroll
    for (int j = 0; j < 8; ++j) {
      int c = c0 + j;
      float e = __expf(sc[r * 336 + (c < 336 ? c : 335)] - m) * inv;
      vals[j] = (c < 336) ? e : 0.f;
    }
    int chunk = c0 >> 5, cw = c0 & 31;
    int lane_d = r + 16 * ((cw >> 4) & 1);
    int j0 = cw & 8;
    store8(&psf[chunk * 512 + lane_d * 16 + j0], vals);
  }
  __syncthreads();

  // pass 2: out = P @ V
  FragC acc[4];
#pragma unroll
  for (int t = 0; t < 4; ++t)
#pragma unroll
    for (int i = 0; i < 8; ++i) acc[t].f[i] = 0.f;

  for (int kt = 0; kt < KLEN; kt += 32) {
    // V tile 32x64: gather along k (lane-coalesced loads, contiguous stores),
    // branch-free via clamp + select
#pragma unroll
    for (int t = 0; t < 8; ++t) {
      int task = lane + t * 32;     // 0..255
      int c  = task & 63;
      int k0 = (task >> 6) * 8;     // 0,8,16,24
      float vals[8];
#pragma unroll
      for (int j = 0; j < 8; ++j) {
        int kr = kt + k0 + j;
        int krc = kr < KLEN ? kr : (KLEN - 1);
        float v = vbuf[(size_t)krc * BD + headoff + c];
        vals[j] = (kr < KLEN) ? v : 0.f;
      }
      int lane_d = (c & 15) + 16 * ((k0 >> 4) & 1);
      store8(&vsf[(c >> 4) * 512 + lane_d * 16 + (k0 & 8)], vals);
    }
    __syncthreads();
    FragB pa = load_frag(psf + (kt >> 5) * 512, lane);
#pragma unroll
    for (int t = 0; t < 4; ++t) {
      FragB vbf = load_frag(vsf + t * 512, lane);
      acc[t].v = wmma_bf16(pa, vbf, acc[t].v);
    }
    __syncthreads();
  }

  const int n = lane & 15;
  const int mrow0 = (lane < 16) ? 0 : 8;
#pragma unroll
  for (int t = 0; t < 4; ++t)
#pragma unroll
    for (int r = 0; r < 8; ++r)
      obuf[(size_t)(q0 + mrow0 + r) * BD + headoff + t * 16 + n] = acc[t].f[r];
}

// ---------------- layernorm: one block (256 thr) per row ----------------
__global__ void layernorm_kernel(const float* __restrict__ x,
                                 const float* __restrict__ g,
                                 const float* __restrict__ be,
                                 float* __restrict__ y, int C) {
  __shared__ float red[256];
  const int row = blockIdx.x;
  const int tid = threadIdx.x;
  const float* xr = x + (size_t)row * C;

  float s = 0.f;
  for (int c = tid; c < C; c += 256) s += xr[c];
  red[tid] = s; __syncthreads();
  for (int o = 128; o > 0; o >>= 1) {
    if (tid < o) red[tid] += red[tid + o];
    __syncthreads();
  }
  float mean = red[0] / (float)C;
  __syncthreads();

  float v = 0.f;
  for (int c = tid; c < C; c += 256) { float d = xr[c] - mean; v += d * d; }
  red[tid] = v; __syncthreads();
  for (int o = 128; o > 0; o >>= 1) {
    if (tid < o) red[tid] += red[tid + o];
    __syncthreads();
  }
  float rstd = rsqrtf(red[0] / (float)C + 1e-5f);

  for (int c = tid; c < C; c += 256)
    y[(size_t)row * C + c] = (xr[c] - mean) * rstd * g[c] + be[c];
}

// ---------------- small helper kernels ----------------
__global__ void reorder_x_kernel(const float* __restrict__ xt,
                                 float* __restrict__ x) {
  int i = blockIdx.x * 256 + threadIdx.x;
  if (i >= TR * BD) return;
  int t = i / BD, rem = i % BD;
  int b = rem / DM, j = rem % DM;
  x[i] = xt[(size_t)b * (TIN * TRI) + (size_t)t * DM + j];
}

__global__ void build_cat_kernel(const float* __restrict__ x,
                                 float* __restrict__ cat) {
  int i = blockIdx.x * 256 + threadIdx.x;
  if (i >= (RT + UU) * BD) return;
  int row = i / BD, rem = i % BD;
  int src;
  if (row < RT) {
    int ii = row >> 2, j = row & 3;
    int st = (ii < NSEG - 1) ? (ii + 1) * 16 : (TR - 4);
    src = st + j;
  } else {
    src = row - RT;
  }
  cat[i] = x[(size_t)src * BD + rem];
}

__global__ void meanpool_kernel(const float* __restrict__ in,
                                float* __restrict__ out, int nseg) {
  int i = blockIdx.x * 256 + threadIdx.x;
  if (i >= nseg * BD) return;
  int seg = i / BD, bd = i % BD;
  float s = 0.f;
#pragma unroll
  for (int t = 0; t < 16; ++t) s += in[(size_t)(seg * 16 + t) * BD + bd];
  out[i] = s * 0.0625f;
}

__global__ void tanh_mems_kernel(const float* __restrict__ outb,
                                 float* __restrict__ mems) {
  int i = blockIdx.x * 256 + threadIdx.x;
  if (i >= (NSEG - 1) * BD) return;
  mems[i] = tanhf(outb[(size_t)(RT + UU) * BD + i]);
}

__global__ void add_kernel(const float* __restrict__ a,
                           const float* __restrict__ b,
                           float* __restrict__ c, int n) {
  int i = blockIdx.x * 256 + threadIdx.x;
  if (i < n) c[i] = a[i] + b[i];
}

__global__ void transpose_final_kernel(const float* __restrict__ cat,
                                       float* __restrict__ tb) {
  int i = blockIdx.x * 256 + threadIdx.x;
  if (i >= NB * UU * DM) return;
  int b = i / (UU * DM);
  int rem = i % (UU * DM);
  int t = rem / DM, d = rem % DM;
  tb[i] = cat[(size_t)(RT + t) * BD + b * DM + d];
}

__global__ void build_mask_kernel(unsigned char* __restrict__ mask) {
  int q = blockIdx.x;
  int k = threadIdx.x;
  if (q >= QLEN || k >= KLEN) return;
  int i; bool is_s = false;
  if (q < RT)            i = q >> 2;
  else if (q < RT + UU)  i = (q - RT) >> 4;
  else                 { i = q - (RT + UU); is_s = true; }
  int ms = max(i - 4, 0);
  int ss = max(16 * i - 32, 0);
  int se = min(16 * (i + 1), UU);
  int w[9] = { ms, i - ms, (NSEG - 1) - i, 4 * i, 4, RT - 4 * i - 4,
               ss, se - ss, UU - se };
  int cum = 0, region = 8;
#pragma unroll
  for (int j = 0; j < 9; ++j) {
    int wj = w[j] > 0 ? w[j] : 0;
    if (k >= cum && k < cum + wj) { region = j; break; }
    cum += wj;
  }
  bool allowed = (region == 4) || (region == 7) || (!is_s && region == 1);
  mask[(size_t)q * KLEN + k] = allowed ? 0 : 1;
}

__global__ void write_lr_kernel(const int* __restrict__ lengths,
                                float* __restrict__ out) {
  int i = threadIdx.x;
  if (i < NB) out[i] = (float)(lengths[i] / 4);
}

// ---------------- host orchestration ----------------
static inline int cdiv(int a, int b) { return (a + b - 1) / b; }

extern "C" void kernel_launch(void* const* d_in, const int* in_sizes, int n_in,
                              void* d_out, int out_size, void* d_ws,
                              size_t ws_size, hipStream_t stream) {
  (void)in_sizes; (void)n_in; (void)out_size; (void)ws_size;

  const float* input    = (const float*)d_in[0];
  const int*   lengths  = (const int*)  d_in[1];
  const float* w_in     = (const float*)d_in[2];
  const float* ln_in_g  = (const float*)d_in[3];
  const float* ln_in_b  = (const float*)d_in[4];
  const float* wq       = (const float*)d_in[5];
  const float* bq       = (const float*)d_in[6];
  const float* wk       = (const float*)d_in[7];
  const float* bk       = (const float*)d_in[8];
  const float* wv       = (const float*)d_in[9];
  const float* bv       = (const float*)d_in[10];
  const float* wo       = (const float*)d_in[11];
  const float* bo       = (const float*)d_in[12];
  const float* ff_ln_g  = (const float*)d_in[13];
  const float* ff_ln_b  = (const float*)d_in[14];
  const float* w1       = (const float*)d_in[15];
  const float* b1       = (const float*)d_in[16];
  const float* w2       = (const float*)d_in[17];
  const float* b2       = (const float*)d_in[18];
  const float* ln_out_g = (const float*)d_in[19];
  const float* ln_out_b = (const float*)d_in[20];
  const float* w_out    = (const float*)d_in[21];
  const float* b_out    = (const float*)d_in[22];
  const float* lng      = (const float*)d_in[23];
  const float* lnb      = (const float*)d_in[24];

  float* wsf = (float*)d_ws;
  size_t off = 0;
  auto alloc = [&](size_t n) { float* p = wsf + off; off += n; return p; };

  float* xt   = alloc((size_t)NB * TIN * TRI);
  float* x    = alloc((size_t)TR * BD);
  float* cat  = alloc((size_t)(RT + UU) * BD);
  float* mems = alloc((size_t)(NSEG - 1) * BD);
  float* qin  = alloc((size_t)QLEN * BD);
  float* kin  = alloc((size_t)KLEN * BD);
  float* qb   = alloc((size_t)QLEN * BD);
  float* kb   = alloc((size_t)KLEN * BD);
  float* vb   = alloc((size_t)KLEN * BD);
  float* att  = alloc((size_t)QLEN * BD);
  float* outb = alloc((size_t)QLEN * BD);
  float* res  = alloc((size_t)(RT + UU) * BD);
  float* hb   = alloc((size_t)(RT + UU) * BD);
  float* h1   = alloc((size_t)(RT + UU) * NB * FFN_ / 8); // 320*2048
  float* tbuf = alloc((size_t)NB * UU * DM);
  float* ybuf = alloc((size_t)NB * UU * OUTD);
  unsigned char* mask = (unsigned char*)(wsf + off);

  const dim3 blk256(256);

  gemm_wmma_kernel<<<dim3(cdiv(NB * TIN, 64), cdiv(TRI, 128)), blk256, 0, stream>>>(
      input, w_in, nullptr, nullptr, xt, NB * TIN, TRI, INF_, 0);

  reorder_x_kernel<<<cdiv(TR * BD, 256), blk256, 0, stream>>>(xt, x);
  build_cat_kernel<<<cdiv((RT + UU) * BD, 256), blk256, 0, stream>>>(x, cat);
  meanpool_kernel<<<cdiv((NSEG - 1) * BD, 256), blk256, 0, stream>>>(x, mems, NSEG - 1);
  build_mask_kernel<<<QLEN, KLEN, 0, stream>>>(mask);

  const int Mq = QLEN * NB;           // 2688
  const int Mk = KLEN * NB;           // 2680
  const int Mc = (RT + UU) * NB;      // 2560

  for (int l = 0; l < NL; ++l) {
    const float* g1  = ln_in_g  + (size_t)l * DM;
    const float* be1 = ln_in_b  + (size_t)l * DM;
    const float* Wq  = wq + (size_t)l * DM * DM;   const float* Bq = bq + (size_t)l * DM;
    const float* Wk  = wk + (size_t)l * DM * DM;   const float* Bk = bk + (size_t)l * DM;
    const float* Wv  = wv + (size_t)l * DM * DM;   const float* Bv = bv + (size_t)l * DM;
    const float* Wo  = wo + (size_t)l * DM * DM;   const float* Bo = bo + (size_t)l * DM;
    const float* g2  = ff_ln_g  + (size_t)l * DM;
    const float* be2 = ff_ln_b  + (size_t)l * DM;
    const float* Wf1 = w1 + (size_t)l * DM * FFN_; const float* Bf1 = b1 + (size_t)l * FFN_;
    const float* Wf2 = w2 + (size_t)l * FFN_ * DM; const float* Bf2 = b2 + (size_t)l * DM;
    const float* g3  = ln_out_g + (size_t)l * DM;
    const float* be3 = ln_out_b + (size_t)l * DM;

    layernorm_kernel<<<Mc, blk256, 0, stream>>>(cat, g1, be1, qin, DM);
    meanpool_kernel<<<cdiv(NSEG * BD, 256), blk256, 0, stream>>>(
        qin + (size_t)RT * BD, qin + (size_t)(RT + UU) * BD, NSEG);
    hipMemcpyAsync(kin, mems, (size_t)(NSEG - 1) * BD * sizeof(float),
                   hipMemcpyDeviceToDevice, stream);
    hipMemcpyAsync(kin + (size_t)(NSEG - 1) * BD, qin,
                   (size_t)(RT + UU) * BD * sizeof(float),
                   hipMemcpyDeviceToDevice, stream);

    gemm_wmma_kernel<<<dim3(cdiv(Mq, 64), cdiv(DM, 128)), blk256, 0, stream>>>(
        qin, Wq, Bq, nullptr, qb, Mq, DM, DM, 0);
    gemm_wmma_kernel<<<dim3(cdiv(Mk, 64), cdiv(DM, 128)), blk256, 0, stream>>>(
        kin, Wk, Bk, nullptr, kb, Mk, DM, DM, 0);
    gemm_wmma_kernel<<<dim3(cdiv(Mk, 64), cdiv(DM, 128)), blk256, 0, stream>>>(
        kin, Wv, Bv, nullptr, vb, Mk, DM, DM, 0);

    attention_kernel<<<dim3(NB * NH, QLEN / 16), dim3(32), 0, stream>>>(
        qb, kb, vb, mask, lengths, att);

    gemm_wmma_kernel<<<dim3(cdiv(Mq, 64), cdiv(DM, 128)), blk256, 0, stream>>>(
        att, Wo, Bo, nullptr, outb, Mq, DM, DM, 0);

    tanh_mems_kernel<<<cdiv((NSEG - 1) * BD, 256), blk256, 0, stream>>>(outb, mems);
    add_kernel<<<cdiv(Mc * DM, 256), blk256, 0, stream>>>(outb, cat, res, Mc * DM);

    layernorm_kernel<<<Mc, blk256, 0, stream>>>(res, g2, be2, hb, DM);
    gemm_wmma_kernel<<<dim3(cdiv(Mc, 64), cdiv(FFN_, 128)), blk256, 0, stream>>>(
        hb, Wf1, Bf1, nullptr, h1, Mc, FFN_, DM, 1);
    gemm_wmma_kernel<<<dim3(cdiv(Mc, 64), cdiv(DM, 128)), blk256, 0, stream>>>(
        h1, Wf2, Bf2, res, res, Mc, DM, FFN_, 0);

    layernorm_kernel<<<Mc, blk256, 0, stream>>>(res, g3, be3, cat, DM);
  }

  transpose_final_kernel<<<cdiv(NB * UU * DM, 256), blk256, 0, stream>>>(cat, tbuf);
  gemm_wmma_kernel<<<dim3(cdiv(NB * UU, 64), cdiv(OUTD, 128)), blk256, 0, stream>>>(
      tbuf, w_out, b_out, nullptr, ybuf, NB * UU, OUTD, DM, 0);
  layernorm_kernel<<<NB * UU, blk256, 0, stream>>>(ybuf, lng, lnb, (float*)d_out, OUTD);

  write_lr_kernel<<<1, 32, 0, stream>>>(lengths,
                                        (float*)d_out + (size_t)NB * UU * OUTD);
}
